// APPNP_35588099015574
// MI455X (gfx1250) — matrix operand order repaired and checked by reading, not compiled, not run on hardware
//
#include <hip/hip_runtime.h>
#include <hip/hip_bf16.h>

// ---------------------------------------------------------------------------
// APPNP GNN on MI455X (gfx1250):
//   h = BN1(relu(x@W1+b1)); h0 = BN2(h@W2+b2); 10x APPNP propagation.
// GEMMs use v_wmma_f32_16x16x32_bf16 with split-bf16 (3 WMMA / k-step) for
// ~fp32 accuracy. Propagation is fp32, L2-resident (192MB L2 >> working set),
// scatter via one wave per edge with float4 gathers + f32 atomics.
// ---------------------------------------------------------------------------

#define NNODES 50000
#define NEDGES 800000
#define FIN    512
#define HDIM   128
#define KPROP  10
#define BN_EPS 1e-5f

typedef __attribute__((ext_vector_type(16))) __bf16 v16bf;
typedef __attribute__((ext_vector_type(8)))  float  v8f;

__device__ __forceinline__ __bf16 bf_hi(float f) { return (__bf16)f; }
__device__ __forceinline__ __bf16 bf_lo(float f) {
  return (__bf16)(f - (float)(__bf16)f);
}

// One wave computes one 16x16 f32 tile of  out = BN(act(A@W + bias)).
// A: [M x LDA] row-major fp32, W: [LDA x 128] row-major fp32.
// Wave-per-tile; block = 256 threads = 8 waves = all 8 N-tiles of one M row.
template <int KSTEPS, int LDA, bool RELU>
__global__ __launch_bounds__(256) void gemm_bn_kernel(
    const float* __restrict__ A, const float* __restrict__ W,
    const float* __restrict__ bias, const float* __restrict__ gamma,
    const float* __restrict__ beta, const float* __restrict__ mean,
    const float* __restrict__ var, float* __restrict__ out) {
  const int lane = threadIdx.x & 31;
  const int wave = threadIdx.x >> 5;
  const int m0 = blockIdx.x * 16;
  const int n0 = wave * 16;

  // A-matrix 16x32 bf16 layout (ISA 7.12.2): lane l holds row m = l&15;
  //   elem i in 0..7  -> k = kb + i        (kb = (l>=16) ? 8 : 0)
  //   elem i in 8..15 -> k = 16 + kb + i-8
  const int mr  = lane & 15;
  const int kbA = (lane >> 4) * 8;
  // B-matrix 32x16 bf16 layout: lane l holds col n = l&15;
  //   elem i in 0..7  -> k = kb2 + i, elem i in 8..15 -> k = 16 + kb2 + i-8  (kb2 = (l>=16) ? 8 : 0)
  const int nc  = lane & 15;
  const int kbB = (lane >> 4) * 8;

  v8f c0 = {};  // ah*bh accumulator
  v8f c1 = {};  // correction terms accumulator

  const float* arow = A + (size_t)(m0 + mr) * LDA + kbA;
  const float* bcol = W + (size_t)kbB * HDIM + n0 + nc;

  for (int ks = 0; ks < KSTEPS; ++ks) {
    v16bf ah, al, bh, bl;
#pragma unroll
    for (int i = 0; i < 8; ++i) {
      float a0 = arow[i];
      float a1 = arow[16 + i];
      ah[i]     = bf_hi(a0);
      al[i]     = bf_lo(a0);
      ah[8 + i] = bf_hi(a1);
      al[8 + i] = bf_lo(a1);
    }
#pragma unroll
    for (int i = 0; i < 16; ++i) {
      float bv = bcol[(size_t)(i < 8 ? i : i + 8) * HDIM];
      bh[i] = bf_hi(bv);
      bl[i] = bf_lo(bv);
    }
    // a*b ~= ah*bh + al*bh + ah*bl  (fp32 accumulate)
    c0 = __builtin_amdgcn_wmma_f32_16x16x32_bf16(false, ah, false, bh,
                                                 (short)0, c0, false, false);
    c1 = __builtin_amdgcn_wmma_f32_16x16x32_bf16(false, al, false, bh,
                                                 (short)0, c1, false, false);
    c1 = __builtin_amdgcn_wmma_f32_16x16x32_bf16(false, ah, false, bl,
                                                 (short)0, c1, false, false);
    arow += 32;
    bcol += (size_t)32 * HDIM;
  }

  // C/D layout: lane l, vgpr r -> (m = r + ((l>=16)?8:0), n = l&15)
  const int n = n0 + nc;
  const float scale = gamma[n] * rsqrtf(var[n] + BN_EPS);
  const float shift = beta[n] - mean[n] * scale;
  const float b = bias[n];
  const int mbase = m0 + ((lane >> 4) * 8);
#pragma unroll
  for (int r = 0; r < 8; ++r) {
    float v = c0[r] + c1[r] + b;
    if (RELU) v = fmaxf(v, 0.0f);
    v = v * scale + shift;
    out[(size_t)(mbase + r) * HDIM + n] = v;
  }
}

// ------------------------- graph preprocessing -----------------------------

__global__ void init_deg_kernel(float* __restrict__ deg) {
  int v = blockIdx.x * blockDim.x + threadIdx.x;
  if (v < NNODES) deg[v] = 1.0f;  // self loop
}

__global__ void edge_prep_kernel(const long long* __restrict__ ei,
                                 int* __restrict__ src, int* __restrict__ dst,
                                 float* __restrict__ deg) {
  int e = blockIdx.x * blockDim.x + threadIdx.x;
  if (e < NEDGES) {
    int s = (int)ei[e];
    int d = (int)ei[(size_t)NEDGES + e];
    src[e] = s;
    dst[e] = d;
    atomicAdd(&deg[d], 1.0f);
  }
}

__global__ void dinv_kernel(const float* __restrict__ deg,
                            float* __restrict__ dinv) {
  int v = blockIdx.x * blockDim.x + threadIdx.x;
  if (v < NNODES) dinv[v] = rsqrtf(fmaxf(deg[v], 1.0f));
}

__global__ void norm_kernel(const int* __restrict__ src,
                            const int* __restrict__ dst,
                            const float* __restrict__ dinv,
                            float* __restrict__ norm) {
  int e = blockIdx.x * blockDim.x + threadIdx.x;
  if (e < NEDGES) norm[e] = dinv[src[e]] * dinv[dst[e]];
}

// -------------------------- propagation kernels ----------------------------

// One wave per edge; each lane handles 4 features (128/32): float4 gather of
// h[src], scale by norm, 4 f32 atomic adds into agg[dst]. All L2-resident.
__global__ __launch_bounds__(256) void scatter_kernel(
    const int* __restrict__ src, const int* __restrict__ dst,
    const float* __restrict__ norm, const float* __restrict__ h,
    float* __restrict__ agg) {
  unsigned int t = blockIdx.x * 256u + threadIdx.x;
  int e = (int)(t >> 5);
  int lane = (int)(t & 31);
  if (e >= NEDGES) return;
  int s = src[e];
  int d = dst[e];
  float w = norm[e];
  const float4 hv = *((const float4*)(h + (size_t)s * HDIM) + lane);
  float* ap = agg + (size_t)d * HDIM + lane * 4;
  atomicAdd(ap + 0, hv.x * w);
  atomicAdd(ap + 1, hv.y * w);
  atomicAdd(ap + 2, hv.z * w);
  atomicAdd(ap + 3, hv.w * w);
}

// h_next = 0.9*(agg + dinv[v]^2 * h_cur) + 0.1*h0 ; reset agg to 0.
__global__ __launch_bounds__(256) void combine_kernel(
    const float* __restrict__ hcur, const float* __restrict__ h0,
    const float* __restrict__ dinv, float* __restrict__ agg,
    float* __restrict__ hnext) {
  int t = blockIdx.x * 256 + threadIdx.x;
  if (t >= NNODES * 32) return;
  int v = t >> 5;
  int q = t & 31;
  float dv = dinv[v];
  float sl = dv * dv;  // self-loop norm
  size_t off = (size_t)v * HDIM + q * 4;
  float4 a  = *(float4*)(agg + off);
  float4 hc = *(const float4*)(hcur + off);
  float4 hz = *(const float4*)(h0 + off);
  float4 r;
  r.x = 0.9f * (a.x + sl * hc.x) + 0.1f * hz.x;
  r.y = 0.9f * (a.y + sl * hc.y) + 0.1f * hz.y;
  r.z = 0.9f * (a.z + sl * hc.z) + 0.1f * hz.z;
  r.w = 0.9f * (a.w + sl * hc.w) + 0.1f * hz.w;
  *(float4*)(hnext + off) = r;
  float4 z = {0.0f, 0.0f, 0.0f, 0.0f};
  *(float4*)(agg + off) = z;  // ready for next iteration's scatter
}

// ------------------------------- launcher ----------------------------------

extern "C" void kernel_launch(void* const* d_in, const int* in_sizes, int n_in,
                              void* d_out, int out_size, void* d_ws,
                              size_t ws_size, hipStream_t stream) {
  const float* x        = (const float*)d_in[0];
  const long long* ei   = (const long long*)d_in[1];
  const float* W1 = (const float*)d_in[2];
  const float* b1 = (const float*)d_in[3];
  const float* g1 = (const float*)d_in[4];
  const float* be1 = (const float*)d_in[5];
  const float* mu1 = (const float*)d_in[6];
  const float* v1 = (const float*)d_in[7];
  const float* W2 = (const float*)d_in[8];
  const float* b2 = (const float*)d_in[9];
  const float* g2 = (const float*)d_in[10];
  const float* be2 = (const float*)d_in[11];
  const float* mu2 = (const float*)d_in[12];
  const float* v2 = (const float*)d_in[13];
  float* out = (float*)d_out;

  // workspace carve-up (256B aligned)
  char* ws = (char*)d_ws;
  size_t off = 0;
  auto carve = [&](size_t bytes) -> char* {
    char* p = ws + off;
    off += (bytes + 255) & ~(size_t)255;
    return p;
  };
  const size_t NH = (size_t)NNODES * HDIM;
  float* h1   = (float*)carve(NH * 4);      // GEMM1 output; reused as agg
  float* h0   = (float*)carve(NH * 4);      // MLP output (teleport vector)
  float* hA   = (float*)carve(NH * 4);
  float* hB   = (float*)carve(NH * 4);
  float* deg  = (float*)carve(NNODES * 4);
  float* dinv = (float*)carve(NNODES * 4);
  float* norm = (float*)carve(NEDGES * 4);
  int*   srcI = (int*)carve(NEDGES * 4);
  int*   dstI = (int*)carve(NEDGES * 4);
  float* agg  = h1;

  dim3 blk(256);

  // Graph preprocessing (independent of GEMMs, same stream = ordered)
  init_deg_kernel<<<(NNODES + 255) / 256, blk, 0, stream>>>(deg);
  edge_prep_kernel<<<(NEDGES + 255) / 256, blk, 0, stream>>>(ei, srcI, dstI, deg);
  dinv_kernel<<<(NNODES + 255) / 256, blk, 0, stream>>>(deg, dinv);
  norm_kernel<<<(NEDGES + 255) / 256, blk, 0, stream>>>(srcI, dstI, dinv, norm);

  // MLP: h1 = BN1(relu(x@W1+b1)); h0 = BN2(h1@W2+b2)
  gemm_bn_kernel<FIN / 32, FIN, true>
      <<<NNODES / 16, blk, 0, stream>>>(x, W1, b1, g1, be1, mu1, v1, h1);
  gemm_bn_kernel<HDIM / 32, HDIM, false>
      <<<NNODES / 16, blk, 0, stream>>>(h1, W2, b2, g2, be2, mu2, v2, h0);

  // h1 consumed; reuse as agg (zero it once, combine re-zeros each iter)
  (void)hipMemsetAsync(agg, 0, NH * 4, stream);

  const float* cur = h0;
  for (int it = 0; it < KPROP; ++it) {
    float* nxt = (it == KPROP - 1) ? out : ((it & 1) ? hB : hA);
    scatter_kernel<<<(NEDGES * 32) / 256, blk, 0, stream>>>(srcI, dstI, norm,
                                                            cur, agg);
    combine_kernel<<<(NNODES * 32 + 255) / 256, blk, 0, stream>>>(cur, h0, dinv,
                                                                  agg, nxt);
    cur = nxt;
  }
}
